// MambaModule_79147657331141
// MI455X (gfx1250) — matrix operand order repaired and checked by reading, not compiled, not run on hardware
//
#include <hip/hip_runtime.h>
#include <math.h>

// ---- problem constants (match reference) ----
#define BATCH 2
#define CCH   256      // d_model
#define LSEQ  1024     // H*W
#define DIN   512      // d_inner
#define NST   16       // d_state
#define DTR   16       // dt_rank
#define BL    (BATCH*LSEQ)   // 2048 rows

typedef __attribute__((ext_vector_type(16))) _Float16 v16h;
typedef __attribute__((ext_vector_type(8)))  float    v8f;

__device__ __forceinline__ float sig_(float x){ return 1.0f/(1.0f+__expf(-x)); }

// ---------------- layout shuffles ----------------
// x (B,C,H,W) -> xt (B*L, C)
__global__ void k_transpose_in(const float* __restrict__ x, float* __restrict__ xt){
  int idx = blockIdx.x*blockDim.x + threadIdx.x;
  if (idx >= BL*CCH) return;
  int c = idx & (CCH-1);
  int bl = idx >> 8;
  int b = bl >> 10, l = bl & (LSEQ-1);
  xt[idx] = x[(b*CCH + c)*LSEQ + l];
}

// channel flip: dst[bl,c] = src[bl, C-1-c]
__global__ void k_flip(const float* __restrict__ src, float* __restrict__ dst){
  int idx = blockIdx.x*blockDim.x + threadIdx.x;
  if (idx >= BL*CCH) return;
  int c = idx & (CCH-1);
  int bl = idx >> 8;
  dst[idx] = src[bl*CCH + (CCH-1-c)];
}

// final: out (B,C,H,W) = transpose(flip(xt))
__global__ void k_final(const float* __restrict__ xt, float* __restrict__ out){
  int idx = blockIdx.x*blockDim.x + threadIdx.x;
  if (idx >= BATCH*CCH*LSEQ) return;
  int l = idx & (LSEQ-1);
  int rest = idx >> 10;
  int c = rest & (CCH-1);
  int b = rest >> 8;
  out[idx] = xt[(b*LSEQ + l)*CCH + (CCH-1-c)];
}

// ---------------- WMMA GEMM ----------------
// C[M,N] = act( A[M,K](row stride LDA) * W[K,N] + bias ), f16 in / f32 acc.
// K, N, LDA compile-time -> all fragment/store offsets become immediates.
// K multiple of 32, or exactly 16 with static zero-padding.
// NT 16-wide output tiles per wave (A-fragment reuse). All loads unconditional.
template<int K, int N, int LDA, int NT, int ACT>
__global__ void k_gemm_wmma(const float* __restrict__ A,
                            const float* __restrict__ W,
                            const float* __restrict__ bias,
                            float* __restrict__ C,
                            int M){
  int gid  = blockIdx.x*blockDim.x + threadIdx.x;
  int wave = gid >> 5;
  int lane = threadIdx.x & 31;
  const int tilesN = N / (16*NT);
  int tilesM = M >> 4;
  if (wave >= tilesM*tilesN) return;          // wave-uniform: EXEC all-ones at WMMA
  int tm = wave / tilesN;
  int tn = wave - tm*tilesN;
  int m_base = tm << 4, n_base = tn * (16*NT);
  int row_a  = m_base + (lane & 15);
  int col0   = n_base + (lane & 15);
  bool hi    = (lane >= 16);
  int  hi8   = hi ? 8 : 0;                    // A K-offset AND C/D row offset
  int  hi16  = hi ? 16 : 0;                   // unused; B K-offset is hi8 too
  const float* arow = A + row_a*LDA;

  v8f acc[NT];
#pragma unroll
  for (int t = 0; t < NT; ++t) acc[t] = (v8f){};

#pragma unroll 4
  for (int K0 = 0; K0 < K; K0 += 32){
    // ---- A fragment: two 8-wide runs of consecutive K -> 4x float4 loads ----
    v16h af;
    const float* ap = arow + K0 + hi8;
    float4 a0 = *(const float4*)(ap);
    float4 a1 = *(const float4*)(ap + 4);
    af[0]=(_Float16)a0.x; af[1]=(_Float16)a0.y; af[2]=(_Float16)a0.z; af[3]=(_Float16)a0.w;
    af[4]=(_Float16)a1.x; af[5]=(_Float16)a1.y; af[6]=(_Float16)a1.z; af[7]=(_Float16)a1.w;
    if constexpr (K != 16){
      float4 a2 = *(const float4*)(ap + 16);
      float4 a3 = *(const float4*)(ap + 20);
      af[8] =(_Float16)a2.x; af[9] =(_Float16)a2.y; af[10]=(_Float16)a2.z; af[11]=(_Float16)a2.w;
      af[12]=(_Float16)a3.x; af[13]=(_Float16)a3.y; af[14]=(_Float16)a3.z; af[15]=(_Float16)a3.w;
    } else {
      // K=16: elements 8..15 address K>=16 -> statically zero
#pragma unroll
      for (int e = 8; e < 16; ++e) af[e] = (_Float16)0.0f;
    }
#pragma unroll
    for (int t = 0; t < NT; ++t){
      int coln = col0 + t*16;
      // one base address, (e + (e & 8))*N immediate offsets (N compile-time)
      const float* wb = W + (K0 + hi8)*N + coln;
      v16h bf;
#pragma unroll
      for (int e = 0; e < 16; ++e)
        bf[e] = (K == 16 && e >= 8) ? (_Float16)0.0f : (_Float16)wb[(e + (e & 8))*N];  // coalesced across lanes
      acc[t] = __builtin_amdgcn_wmma_f32_16x16x32_f16(false, af, false, bf,
                                                      (short)0, acc[t], false, false);
    }
  }
#pragma unroll
  for (int t = 0; t < NT; ++t){
    int coln = col0 + t*16;
    float bv = bias ? bias[coln] : 0.0f;
    float* cb = C + (m_base + hi8)*N + coln;     // r*N immediate offsets
#pragma unroll
    for (int r = 0; r < 8; ++r){
      float val = acc[t][r] + bv;
      if constexpr (ACT == 1)
        val = (val > 20.0f) ? val : log1pf(__expf(val));  // softplus
      cb[r*N] = val;
    }
  }
}

// ---------------- depthwise causal conv + SiLU ----------------
__global__ void k_conv_silu(const float* __restrict__ xr, const float* __restrict__ cw,
                            const float* __restrict__ cb, float* __restrict__ xs){
  int idx = blockIdx.x*blockDim.x + threadIdx.x;
  if (idx >= BL*DIN) return;
  int d  = idx & (DIN-1);
  int bl = idx >> 9;
  int b = bl >> 10, l = bl & (LSEQ-1);
  float acc = cb[d];
#pragma unroll
  for (int j = 0; j < 4; ++j){
    int ll = l - 3 + j;
    if (ll >= 0) acc += cw[d*4 + j] * xr[(b*LSEQ + ll)*(2*DIN) + d];
  }
  xs[idx] = acc * sig_(acc);                 // SiLU
}

// ---------------- selective scan ----------------
// one thread per (b,d); 16 states in registers; serial over L.
__global__ void k_scan(const float* __restrict__ xs, const float* __restrict__ xdbl,
                       const float* __restrict__ delta, const float* __restrict__ xr,
                       const float* __restrict__ A_log, const float* __restrict__ Dp,
                       float* __restrict__ y){
  int t = blockIdx.x*blockDim.x + threadIdx.x;
  if (t >= BATCH*DIN) return;
  int b = t >> 9, d = t & (DIN-1);
  float h[NST], Ar[NST];
#pragma unroll
  for (int n = 0; n < NST; ++n){ h[n] = 0.0f; Ar[n] = -__expf(A_log[d*NST + n]); }
  float Dd = Dp[d];
  for (int l = 0; l < LSEQ; ++l){
    int bl = b*LSEQ + l;
    __builtin_prefetch(xdbl + (bl + 8)*(DTR + 2*NST), 0, 1);  // cover serial latency
    float dl = delta[bl*DIN + d];
    float xv = xs[bl*DIN + d];
    // B_t (16f) and C_t (16f) as 8x float4 (16B-aligned: bl*48+16)
    const float4* xb4 = (const float4*)(xdbl + bl*(DTR + 2*NST) + DTR);
    float4 b4[4], c4[4];
#pragma unroll
    for (int q = 0; q < 4; ++q){ b4[q] = xb4[q]; c4[q] = xb4[q + 4]; }
    const float* Bn = (const float*)b4;
    const float* Cn = (const float*)c4;
    float acc = 0.0f;
#pragma unroll
    for (int n = 0; n < NST; ++n){
      float dA = __expf(dl * Ar[n]);
      h[n] = dA*h[n] + dl * Bn[n] * xv;
      acc += h[n] * Cn[n];
    }
    acc += xv * Dd;
    float res = xr[bl*(2*DIN) + DIN + d];       // gate branch
    y[bl*DIN + d] = acc * (res * sig_(res));    // y * silu(res)
  }
}

// ---------------- LayerNorm + residual (in-place on xt) ----------------
__global__ void k_ln_residual(const float* __restrict__ mout,
                              const float* __restrict__ w, const float* __restrict__ b,
                              float* __restrict__ xt){
  __shared__ float red[CCH];
  int bl = blockIdx.x, c = threadIdx.x;
  float v = mout[bl*CCH + c];
  red[c] = v; __syncthreads();
  for (int s = CCH/2; s > 0; s >>= 1){ if (c < s) red[c] += red[c+s]; __syncthreads(); }
  float mean = red[0] * (1.0f/CCH); __syncthreads();
  float dv = v - mean;
  red[c] = dv*dv; __syncthreads();
  for (int s = CCH/2; s > 0; s >>= 1){ if (c < s) red[c] += red[c+s]; __syncthreads(); }
  float var = red[0] * (1.0f/CCH);
  xt[bl*CCH + c] += dv * rsqrtf(var + 1e-5f) * w[c] + b[c];
}

// ---------------- host side ----------------
static void run_mamba(hipStream_t s, float* xt,
                      const float* in_w, const float* in_b,
                      const float* conv_w, const float* conv_b,
                      const float* xproj_w, const float* dt_w, const float* dt_b,
                      const float* A_log, const float* Dp,
                      const float* out_w, const float* out_b,
                      const float* ln_w, const float* ln_b,
                      float* xr, float* xs, float* xdbl, float* delta,
                      float* yb, float* mout){
  dim3 T(256);
  // in_proj: 2048x1024x256 -> xr        (NT=2: 128*32 waves)
  k_gemm_wmma<256, 2*DIN, CCH, 2, 0><<<512, T, 0, s>>>(xt, in_w, in_b, xr, BL);
  k_conv_silu<<<(BL*DIN)/256, T, 0, s>>>(xr, conv_w, conv_b, xs);
  // x_proj: 2048x48x512 -> xdbl         (N=48 -> NT=1: 128*3 waves)
  k_gemm_wmma<512, DTR+2*NST, DIN, 1, 0><<<48, T, 0, s>>>(xs, xproj_w, nullptr, xdbl, BL);
  // dt_proj + softplus: 2048x512x16     (K=16 statically padded; 128*16 waves)
  k_gemm_wmma<16, DIN, DTR+2*NST, 2, 1><<<256, T, 0, s>>>(xdbl, dt_w, dt_b, delta, BL);
  k_scan<<<(BATCH*DIN)/256, T, 0, s>>>(xs, xdbl, delta, xr, A_log, Dp, yb);
  // out_proj: 2048x256x512 -> mout      (128*8 waves)
  k_gemm_wmma<512, CCH, DIN, 2, 0><<<128, T, 0, s>>>(yb, out_w, out_b, mout, BL);
  k_ln_residual<<<BL, T, 0, s>>>(mout, ln_w, ln_b, xt);
}

extern "C" void kernel_launch(void* const* d_in, const int* in_sizes, int n_in,
                              void* d_out, int out_size, void* d_ws, size_t ws_size,
                              hipStream_t stream){
  (void)in_sizes; (void)n_in; (void)out_size; (void)ws_size;
  const float* x     = (const float*)d_in[0];
  const float* ln1_w = (const float*)d_in[1];
  const float* ln1_b = (const float*)d_in[2];
  const float* ln2_w = (const float*)d_in[3];
  const float* ln2_b = (const float*)d_in[4];
  const float* p1[11]; const float* p2[11];
  for (int i = 0; i < 11; ++i){ p1[i] = (const float*)d_in[5+i]; p2[i] = (const float*)d_in[16+i]; }

  float* ws = (float*)d_ws;
  float* xt    = ws;                 // 524288  : (B*L, C) working activation
  float* xtb   = ws + 524288;        // 524288  : flipped activation
  float* xr    = ws + 1048576;       // 2097152 : in_proj out (B*L, 2*DIN)
  float* xs    = ws + 3145728;       // 1048576 : conv+silu out (B*L, DIN)
  float* xdbl  = ws + 4194304;       // 98304   : (B*L, 48)
  float* delta = ws + 4292608;       // 1048576 : (B*L, DIN)
  float* yb    = ws + 5341184;       // 1048576 : scan out (B*L, DIN)
  float* mout  = ws + 6389760;       // 524288  : mamba-block out (B*L, C)

  k_transpose_in<<<(BL*CCH)/256, 256, 0, stream>>>(x, xt);

  run_mamba(stream, xt,  p1[0],p1[1],p1[2],p1[3],p1[4],p1[5],p1[6],p1[7],p1[8],p1[9],p1[10],
            ln1_w, ln1_b, xr, xs, xdbl, delta, yb, mout);

  k_flip<<<(BL*CCH)/256, 256, 0, stream>>>(xt, xtb);

  run_mamba(stream, xtb, p2[0],p2[1],p2[2],p2[3],p2[4],p2[5],p2[6],p2[7],p2[8],p2[9],p2[10],
            ln2_w, ln2_b, xr, xs, xdbl, delta, yb, mout);

  k_final<<<(BATCH*CCH*LSEQ)/256, 256, 0, stream>>>(xtb, (float*)d_out);
}
